// FastFeedForward_26233660244167
// MI455X (gfx1250) — matrix operand order repaired and checked by reading, not compiled, not run on hardware
//
#include <hip/hip_runtime.h>
#include <math.h>

// Problem constants (match reference)
#define NTOK        32768      // B*S = 8*4096
#define DIM         512
#define HID         2048
#define NEXP        16
#define TM          64         // tokens per FFN block
#define KC          64         // hidden chunk width
#define NCHUNK      (HID / KC)        // 32
#define TILES_PER_E (NTOK / TM)       // 512

typedef __attribute__((ext_vector_type(16))) __bf16 v16bf;
typedef __attribute__((ext_vector_type(8)))  __bf16 v8bf;
typedef __attribute__((ext_vector_type(8)))  float  v8f;
typedef __attribute__((ext_vector_type(4)))  float  v4f;

// ---------------------------------------------------------------------------
// Fragment loaders (CDNA5 wave32 WMMA layouts, ISA 7.12.2)
// A 16x32 bf16: lanes 0-15 -> M=lane, K runs {0..7,16..23}; lanes 16-31 -> K runs {8..15,24..31}
// B 32x16 bf16: lanes 0-15 -> N=lane, K runs {0..7,16..23}; lanes 16-31 -> K runs {8..15,24..31}
// C/D 16x16 f32: lane n = lane&15, VGPR r -> M = r + 8*(lane>>4)
// ---------------------------------------------------------------------------

__device__ __forceinline__ v16bf load_a_frag_lds(const __bf16* H, int stride,
                                                 int m0, int kk, int lane) {
  int m  = m0 + (lane & 15);
  int kh = (lane & 16) ? 8 : 0;
  v8bf lo = *(const v8bf*)(H + m * stride + kk + kh);
  v8bf hi = *(const v8bf*)(H + m * stride + kk + 16 + kh);
  v16bf r;
#pragma unroll
  for (int i = 0; i < 8; ++i) { r[i] = lo[i]; r[8 + i] = hi[i]; }
  return r;
}

__device__ __forceinline__ v16bf load_b_frag_lds(const __bf16* W, int stride,
                                                 int n0, int kk, int lane) {
  int n  = n0 + (lane & 15);
  int kb = kk + ((lane & 16) ? 8 : 0);
  const v8bf* p = (const v8bf*)(W + n * stride + kb);
  v8bf lo = p[0], hi = p[2];
  v16bf r;
#pragma unroll
  for (int i = 0; i < 8; ++i) { r[i] = lo[i]; r[8 + i] = hi[i]; }
  return r;
}

__device__ __forceinline__ v8f wmma_bf16(v16bf a, v16bf b, v8f c) {
  return __builtin_amdgcn_wmma_f32_16x16x32_bf16(false, a, false, b, (short)0, c,
                                                 false, false);
}

__device__ __forceinline__ float gelu_exact(float v) {
  return 0.5f * v * (1.0f + erff(v * 0.70710678118654752f));
}

// ---------------------------------------------------------------------------
// Kernel 1: binary-tree routing. One wave32 per token.
// ---------------------------------------------------------------------------
__global__ __launch_bounds__(256)
void route_kernel(const float* __restrict__ x, const float* __restrict__ Wr,
                  const float* __restrict__ br, int* __restrict__ leaf,
                  int* __restrict__ counts) {
  int wave = threadIdx.x >> 5;
  int lane = threadIdx.x & 31;
  int t = blockIdx.x * 8 + wave;   // NTOK divisible by 8

  const float* xp = x + (size_t)t * DIM;
  float xv[16];
#pragma unroll
  for (int i = 0; i < 16; ++i) xv[i] = xp[lane + 32 * i];

  int node = 0;
#pragma unroll
  for (int d = 0; d < 4; ++d) {
    int row = (1 << d) - 1 + node;
    const float* wr = Wr + (size_t)row * DIM;
    float p = 0.0f;
#pragma unroll
    for (int i = 0; i < 16; ++i) p += xv[i] * wr[lane + 32 * i];
#pragma unroll
    for (int off = 16; off > 0; off >>= 1) p += __shfl_xor(p, off, 32);
    float s = p + br[row];
    node = node * 2 + (s > 0.0f ? 1 : 0);
  }
  if (lane == 0) {
    leaf[t] = node;
    atomicAdd(&counts[node], 1);
  }
}

// ---------------------------------------------------------------------------
// Kernel 2: 16-entry exclusive prefix sum.
// ---------------------------------------------------------------------------
__global__ void prefix_kernel(const int* __restrict__ counts, int* __restrict__ offsets) {
  if (threadIdx.x == 0) {
    int s = 0;
#pragma unroll
    for (int e = 0; e < NEXP; ++e) { offsets[e] = s; s += counts[e]; }
    offsets[NEXP] = s;
  }
}

// ---------------------------------------------------------------------------
// Kernel 3: scatter token ids into per-expert contiguous lists.
// ---------------------------------------------------------------------------
__global__ __launch_bounds__(256)
void scatter_kernel(const int* __restrict__ leaf, const int* __restrict__ offsets,
                    int* __restrict__ cursor, int* __restrict__ perm) {
  int t = blockIdx.x * 256 + threadIdx.x;
  int e = leaf[t];
  int pos = atomicAdd(&cursor[e], 1);
  perm[offsets[e] + pos] = t;
}

// ---------------------------------------------------------------------------
// Kernel 4: fused expert FFN  y = gelu(x@W1+b1)@W2+b2 for gathered tokens.
// Block: 256 threads (8 waves). 64 tokens/block, one expert/block.
// X tile and weight chunks staged in LDS as bf16; Y accumulated in f32 VGPRs.
// GEMM2 mapped 2 row-tiles x 8 col-tiles per wave so each B fragment feeds
// two WMMAs; inner loops software-pipelined to hide ds_load latency.
// ---------------------------------------------------------------------------
__global__ __launch_bounds__(256, 1)
void ffn_kernel(const float* __restrict__ x,  const float* __restrict__ W1,
                const float* __restrict__ b1, const float* __restrict__ W2,
                const float* __restrict__ b2, const int* __restrict__ offsets,
                const int* __restrict__ perm, float* __restrict__ out) {
  __shared__ int tokIdx[TM];
  __shared__ __align__(16) __bf16 Xs[TM * DIM];   // 64KB: x tile   [m64][k512]
  __shared__ __align__(16) __bf16 Wb[KC * DIM];   // 64KB: W1 chunk [n64][k512] / W2 chunk [n512][k64]
  __shared__ __align__(16) __bf16 Hs[TM * KC];    // 8KB:  gelu(h)  [m64][k64]

  int e    = blockIdx.x >> 9;            // / TILES_PER_E
  int tile = blockIdx.x & (TILES_PER_E - 1);
  int begin = offsets[e];
  int cnt   = offsets[e + 1] - begin;
  int base  = tile * TM;
  if (base >= cnt) return;               // uniform early exit

  int tid  = threadIdx.x;
  if (tid < TM) tokIdx[tid] = (base + tid < cnt) ? perm[begin + base + tid] : -1;
  __syncthreads();

  // Stage X tile once: Xs[r*512+k] = bf16(x[tok[r]*512+k]) (coalesced in k)
  for (int i = tid; i < TM * DIM; i += 256) {
    int r = i >> 9, k = i & 511;
    int tk = tokIdx[r];
    Xs[i] = (tk >= 0) ? (__bf16)x[(size_t)tk * DIM + k] : (__bf16)0.0f;
  }

  int wave = tid >> 5, lane = tid & 31;
  int rt   = wave & 3;    // GEMM1 row tile
  int cg   = wave >> 2;   // GEMM1 column group (2 tiles)
  int rtg  = wave & 1;    // GEMM2 row-tile pair (rows rtg*32..+31)
  int cgrp = wave >> 1;   // GEMM2 column group (8 tiles = 128 cols)
  int rowoff = (lane & 16) ? 8 : 0;

  const float* W1e = W1 + (size_t)e * DIM * HID;
  const float* W2e = W2 + (size_t)e * HID * DIM;

  v8f zero = {0.f, 0.f, 0.f, 0.f, 0.f, 0.f, 0.f, 0.f};
  v8f yacc[16];   // [jj*2 + rtl]
#pragma unroll
  for (int j = 0; j < 16; ++j) yacc[j] = zero;

  // Prefetch first W1 chunk while Xs staging settles (L2 -> WGP).
  for (int l = tid; l < 1024; l += 256) {       // 512 rows x 2 x 128B lines
    int k = l >> 1, off = (l & 1) * 32;
    __builtin_prefetch(W1e + (size_t)k * HID + off, 0, 3);
  }

  for (int c = 0; c < NCHUNK; ++c) {
    __syncthreads();  // previous GEMM2 done reading Wb (and Xs staged on c==0)
    // Stage W1 chunk transposed: Wb[n*512 + k] = bf16(W1e[k*HID + c*64 + n])
    for (int i = tid; i < KC * DIM; i += 256) {
      int k = i >> 6, n = i & 63;
      Wb[n * DIM + k] = (__bf16)W1e[(size_t)k * HID + c * KC + n];
    }
    __syncthreads();

    // Prefetch this chunk's W2 rows (used after GEMM1): 64 rows x 16 x 128B
    for (int l = tid; l < 1024; l += 256) {
      int k = l >> 4, off = (l & 15) * 32;
      __builtin_prefetch(W2e + (size_t)(c * KC + k) * DIM + off, 0, 3);
    }

    // GEMM1: Hc = gelu(X @ W1c + b1c). One A-frag feeds both of this wave's
    // B tiles; fragments prefetched one K-step ahead.
    {
      int jt0 = cg * 2;
      v8f acc0 = zero, acc1 = zero;
      v16bf a   = load_a_frag_lds(Xs, DIM, rt * 16, 0, lane);
      v16bf bb0 = load_b_frag_lds(Wb, DIM, jt0 * 16, 0, lane);
      v16bf bb1 = load_b_frag_lds(Wb, DIM, (jt0 + 1) * 16, 0, lane);
#pragma unroll
      for (int f = 0; f < 16; ++f) {
        v16bf na, nb0, nb1;
        if (f < 15) {
          na  = load_a_frag_lds(Xs, DIM, rt * 16, (f + 1) * 32, lane);
          nb0 = load_b_frag_lds(Wb, DIM, jt0 * 16, (f + 1) * 32, lane);
          nb1 = load_b_frag_lds(Wb, DIM, (jt0 + 1) * 16, (f + 1) * 32, lane);
        }
        acc0 = wmma_bf16(a, bb0, acc0);
        acc1 = wmma_bf16(a, bb1, acc1);
        if (f < 15) { a = na; bb0 = nb0; bb1 = nb1; }
      }
      float bias0 = b1[e * HID + c * KC + jt0 * 16 + (lane & 15)];
      float bias1 = b1[e * HID + c * KC + (jt0 + 1) * 16 + (lane & 15)];
#pragma unroll
      for (int r = 0; r < 8; ++r) {
        int mrow = (rt * 16 + r + rowoff) * KC + (lane & 15);
        Hs[mrow + jt0 * 16]       = (__bf16)gelu_exact(acc0[r] + bias0);
        Hs[mrow + (jt0 + 1) * 16] = (__bf16)gelu_exact(acc1[r] + bias1);
      }
    }
    __syncthreads();  // Hs ready, Wb reads done

    // Stage W2 chunk transposed: Wb[n*64 + k] = bf16(W2e[(c*64+k)*DIM + n])
    for (int i = tid; i < KC * DIM; i += 256) {
      int k = i >> 9, n = i & 511;
      Wb[n * KC + k] = (__bf16)W2e[(size_t)(c * KC + k) * DIM + n];
    }
    __syncthreads();

    // Prefetch next chunk's W1 rows (used after GEMM2)
    if (c + 1 < NCHUNK) {
      for (int l = tid; l < 1024; l += 256) {
        int k = l >> 1, off = (l & 1) * 32;
        __builtin_prefetch(W1e + (size_t)k * HID + (c + 1) * KC + off, 0, 3);
      }
    }

    // GEMM2: Yacc += Hc @ W2c. 2 row-tiles x 8 col-tiles per wave:
    // each B fragment feeds 2 WMMAs, A fragments reused across all 8 tiles.
    {
      v16bf ha[4];
      ha[0] = load_a_frag_lds(Hs, KC, (rtg * 2 + 0) * 16, 0, lane);
      ha[1] = load_a_frag_lds(Hs, KC, (rtg * 2 + 0) * 16, 32, lane);
      ha[2] = load_a_frag_lds(Hs, KC, (rtg * 2 + 1) * 16, 0, lane);
      ha[3] = load_a_frag_lds(Hs, KC, (rtg * 2 + 1) * 16, 32, lane);
      v16bf b0 = load_b_frag_lds(Wb, KC, cgrp * 128, 0, lane);
      v16bf b1f = load_b_frag_lds(Wb, KC, cgrp * 128, 32, lane);
#pragma unroll
      for (int jj = 0; jj < 8; ++jj) {
        v16bf nb0, nb1;
        if (jj < 7) {
          int n0 = (cgrp * 8 + jj + 1) * 16;
          nb0 = load_b_frag_lds(Wb, KC, n0, 0, lane);
          nb1 = load_b_frag_lds(Wb, KC, n0, 32, lane);
        }
        yacc[2 * jj]     = wmma_bf16(ha[0], b0,  yacc[2 * jj]);
        yacc[2 * jj]     = wmma_bf16(ha[1], b1f, yacc[2 * jj]);
        yacc[2 * jj + 1] = wmma_bf16(ha[2], b0,  yacc[2 * jj + 1]);
        yacc[2 * jj + 1] = wmma_bf16(ha[3], b1f, yacc[2 * jj + 1]);
        if (jj < 7) { b0 = nb0; b1f = nb1; }
      }
    }
  }

  // Epilogue: add b2, scatter rows back by token id
#pragma unroll
  for (int jj = 0; jj < 8; ++jj) {
    int col = (cgrp * 8 + jj) * 16 + (lane & 15);
    float bb = b2[e * DIM + col];
#pragma unroll
    for (int rtl = 0; rtl < 2; ++rtl) {
      v8f yv = yacc[2 * jj + rtl];
#pragma unroll
      for (int r = 0; r < 8; ++r) {
        int ml = (rtg * 2 + rtl) * 16 + r + rowoff;
        int tk = tokIdx[ml];
        if (tk >= 0) out[(size_t)tk * DIM + col] = yv[r] + bb;
      }
    }
  }
}

// ---------------------------------------------------------------------------
// Host launcher
// ---------------------------------------------------------------------------
extern "C" void kernel_launch(void* const* d_in, const int* in_sizes, int n_in,
                              void* d_out, int out_size, void* d_ws, size_t ws_size,
                              hipStream_t stream) {
  const float* x  = (const float*)d_in[0];
  const float* Wr = (const float*)d_in[1];
  const float* br = (const float*)d_in[2];
  const float* W1 = (const float*)d_in[3];
  const float* b1 = (const float*)d_in[4];
  const float* W2 = (const float*)d_in[5];
  const float* b2 = (const float*)d_in[6];
  float* out = (float*)d_out;

  char* ws = (char*)d_ws;
  int* leaf    = (int*)ws;                     // NTOK
  int* counts  = leaf + NTOK;                  // 16
  int* offsets = counts + NEXP;                // 17
  int* cursor  = offsets + (NEXP + 1);         // 16
  int* perm    = cursor + NEXP;                // NTOK

  (void)in_sizes; (void)n_in; (void)out_size; (void)ws_size;

  // zero counts + offsets + cursor (contiguous)
  hipMemsetAsync(counts, 0, (NEXP + (NEXP + 1) + NEXP) * sizeof(int), stream);

  route_kernel<<<NTOK / 8, 256, 0, stream>>>(x, Wr, br, leaf, counts);
  prefix_kernel<<<1, 32, 0, stream>>>(counts, offsets);
  scatter_kernel<<<NTOK / 256, 256, 0, stream>>>(leaf, offsets, cursor, perm);
  ffn_kernel<<<NEXP * TILES_PER_E, 256, 0, stream>>>(x, W1, b1, W2, b2,
                                                     offsets, perm, out);
}
